// ODEIntegrator_76493367542065
// MI455X (gfx1250) — matrix operand order repaired and checked by reading, not compiled, not run on hardware
//
#include <hip/hip_runtime.h>
#include <math.h>

#define BATCH 64
#define DIM 256
#define NSTEPS 20
#define MROWS (BATCH * DIM) /* 16384 rows of the state matrix */

typedef __attribute__((ext_vector_type(16))) __bf16 v16bf;
typedef __attribute__((ext_vector_type(2))) __bf16 v2bf;
typedef __attribute__((ext_vector_type(8))) float v8f;

// f32x2 -> packed bf16x2, RNE. Prefer the hardware V_CVT_PK_BF16_F32.
static __device__ __forceinline__ v2bf cvtpk_bf16(float lo, float hi) {
#if defined(__has_builtin) && __has_builtin(__builtin_amdgcn_cvt_pk_bf16_f32)
    return __builtin_amdgcn_cvt_pk_bf16_f32(lo, hi);
#else
    v2bf r;
    r[0] = (__bf16)lo;
    r[1] = (__bf16)hi;
    return r;
#endif
}

// Hardware tanh if available (V_TANH_F32 on gfx1250), else libm.
static __device__ __forceinline__ float fast_tanh(float x) {
#if defined(__has_builtin) && __has_builtin(__builtin_amdgcn_tanhf)
    return __builtin_amdgcn_tanhf(x);
#else
    return tanhf(x);
#endif
}

// ---------------------------------------------------------------------------
// Initial state: v[b*DIM + k][j] = dot(z[b,0,j,k,:], W_in[j,k,:]) + b_in[j,k]
// (einsum "bojkc,jkc->bojk" followed by swapaxes(-1,-2)).
// 16x16 LDS tile transpose so both the z/W reads and the v writes coalesce.
// ---------------------------------------------------------------------------
__global__ __launch_bounds__(256) void init_v0_kernel(
    const float* __restrict__ z, const float* __restrict__ Win,
    const float* __restrict__ bin, float* __restrict__ v) {
    __shared__ float tile[16][17];
    const int tx = threadIdx.x, ty = threadIdx.y;
    const int jt = blockIdx.x, kt = blockIdx.y, b = blockIdx.z;
    const int j = jt * 16 + ty;
    const int k = kt * 16 + tx;
    const float2 zz = *(const float2*)(z + (((size_t)b * DIM + j) * DIM + k) * 2);
    const float2 ww = *(const float2*)(Win + ((size_t)j * DIM + k) * 2);
    tile[ty][tx] = zz.x * ww.x + zz.y * ww.y + bin[(size_t)j * DIM + k];
    __syncthreads();
    // write v[b, k, j]; element (j = jt*16+tx, k = kt*16+ty) lives in tile[tx][ty]
    const int row = b * DIM + kt * 16 + ty;
    const int col = jt * 16 + tx;
    v[(size_t)row * DIM + col] = tile[tx][ty];
}

// ---------------------------------------------------------------------------
// Pack Wf (f32, K=256 x N=256, row-major) into the WMMA bf16 B-operand layout.
// dword index = ((kc*16 + nt)*32 + lane)*8 + d
//   lane 0-15 : n = nt*16 + lane,      K = kc*32 + 2d, 2d+1 (d<4); kc*32 + 16 + 2(d-4), +1 (d>=4)
//   lane16-31 : n = nt*16 + lane-16,   K = kc*32 + 8 + 2d, +1 (d<4); kc*32 + 24 + 2(d-4), +1 (d>=4)
// ---------------------------------------------------------------------------
__global__ __launch_bounds__(256) void pack_wf_kernel(
    const float* __restrict__ Wf, unsigned* __restrict__ WfB) {
    const int tid = blockIdx.x * 256 + threadIdx.x; // 0..32767
    const int d = tid & 7;
    const int lane = (tid >> 3) & 31;
    const int nt = (tid >> 8) & 15;
    const int kc = tid >> 12;
    const int n = nt * 16 + (lane & 15);
    const int k0 = kc * 32 + ((d >> 2) << 4) + ((lane >> 4) << 3) + 2 * (d & 3);
    union { v2bf p; unsigned u; } cv;
    cv.p = cvtpk_bf16(Wf[(size_t)k0 * DIM + n], Wf[(size_t)(k0 + 1) * DIM + n]);
    WfB[tid] = cv.u;
}

// ---------------------------------------------------------------------------
// One RK4 stage: K = tanh((v + coef*kin) @ Wf)  via bf16 WMMA, f32 accumulate.
//   STAGE 0     : kout = K;  accb  = wgt*K
//   STAGE 1/2   : kout = K;  accb += wgt*K
//   STAGE 3     : vout = v + accb + wgt*K
// Block = 256 threads = 8 waves. Wave w: rowTile = blockIdx.x*4 + w/2 (16 rows),
// column half = w&1 (128 cols, 8 WMMA tiles). Wf staged via 64KB LDS in 2 phases.
// B fragments explicitly double-buffered so ds_load overlaps the matrix op.
// ---------------------------------------------------------------------------
template <int STAGE>
__global__ __launch_bounds__(256) void rk4_stage_kernel(
    const float* __restrict__ v, const float* __restrict__ kin,
    float* __restrict__ kout, float* __restrict__ accb,
    const uint4* __restrict__ gB4, float* __restrict__ vout,
    float coef, float wgt) {
    __shared__ uint4 sB4[4096]; // 64 KB = half of packed Wf

    const int t = threadIdx.x;
    const int lane = t & 31;
    const int wv = t >> 5;
    const int rowBase = (blockIdx.x * 4 + (wv >> 1)) * 16;
    const int nhalf = wv & 1;
    const int hf = lane >> 4;
    const int arow = rowBase + (lane & 15);

    const float* __restrict__ vr = v + (size_t)arow * DIM;
    const float* __restrict__ kr = (STAGE > 0) ? (kin + (size_t)arow * DIM) : nullptr;

    v8f acc[8];
#pragma unroll
    for (int i = 0; i < 8; ++i) acc[i] = (v8f){0.f, 0.f, 0.f, 0.f, 0.f, 0.f, 0.f, 0.f};

    union BF { v16bf bv; uint4 q[2]; };

#pragma unroll
    for (int kc = 0; kc < 8; ++kc) {
        if ((kc & 3) == 0) { // stage half of packed Wf into LDS
            const int p = kc >> 2;
            __syncthreads();
#pragma unroll
            for (int it = 0; it < 16; ++it)
                sB4[t + it * 256] = gB4[p * 4096 + t + it * 256];
            __syncthreads();
        }
        // ---- A fragment: 16 bf16 along K per lane, per ISA 16-bit A layout.
        const int i0 = kc * 32 + hf * 8;
        float a[16];
        {
            float4 x0 = *(const float4*)(vr + i0);
            float4 x1 = *(const float4*)(vr + i0 + 4);
            float4 x2 = *(const float4*)(vr + i0 + 16);
            float4 x3 = *(const float4*)(vr + i0 + 20);
            a[0] = x0.x;  a[1] = x0.y;  a[2] = x0.z;  a[3] = x0.w;
            a[4] = x1.x;  a[5] = x1.y;  a[6] = x1.z;  a[7] = x1.w;
            a[8] = x2.x;  a[9] = x2.y;  a[10] = x2.z; a[11] = x2.w;
            a[12] = x3.x; a[13] = x3.y; a[14] = x3.z; a[15] = x3.w;
            if (STAGE > 0) {
                float4 y0 = *(const float4*)(kr + i0);
                float4 y1 = *(const float4*)(kr + i0 + 4);
                float4 y2 = *(const float4*)(kr + i0 + 16);
                float4 y3 = *(const float4*)(kr + i0 + 20);
                a[0] = fmaf(coef, y0.x, a[0]);   a[1] = fmaf(coef, y0.y, a[1]);
                a[2] = fmaf(coef, y0.z, a[2]);   a[3] = fmaf(coef, y0.w, a[3]);
                a[4] = fmaf(coef, y1.x, a[4]);   a[5] = fmaf(coef, y1.y, a[5]);
                a[6] = fmaf(coef, y1.z, a[6]);   a[7] = fmaf(coef, y1.w, a[7]);
                a[8] = fmaf(coef, y2.x, a[8]);   a[9] = fmaf(coef, y2.y, a[9]);
                a[10] = fmaf(coef, y2.z, a[10]); a[11] = fmaf(coef, y2.w, a[11]);
                a[12] = fmaf(coef, y3.x, a[12]); a[13] = fmaf(coef, y3.y, a[13]);
                a[14] = fmaf(coef, y3.z, a[14]); a[15] = fmaf(coef, y3.w, a[15]);
            }
        }
        union { v16bf bv; v2bf p2[8]; } A;
#pragma unroll
        for (int dd = 0; dd < 8; ++dd) A.p2[dd] = cvtpk_bf16(a[2 * dd], a[2 * dd + 1]);

        // ---- inner N loop, B double-buffered from LDS
        const uint4* bp = &sB4[(kc & 3) * 1024 + (nhalf * 8) * 64 + lane * 2];
        BF bb[2];
        bb[0].q[0] = bp[0];
        bb[0].q[1] = bp[1];
#pragma unroll
        for (int ntl = 0; ntl < 8; ++ntl) {
            const int cur = ntl & 1;
            if (ntl < 7) {
                bb[cur ^ 1].q[0] = bp[(ntl + 1) * 64];
                bb[cur ^ 1].q[1] = bp[(ntl + 1) * 64 + 1];
            }
            acc[ntl] = __builtin_amdgcn_wmma_f32_16x16x32_bf16(
                false, A.bv, false, bb[cur].bv, (short)0, acc[ntl], false, false);
        }
    }

    // Stage 3 writes v while sibling waves of this block may still read it as A.
    __syncthreads();

    // Epilogue. C/D layout: lanes 0-15: VGPR r -> (M=r, N=lane);
    //                       lanes16-31: VGPR r -> (M=8+r, N=lane-16).
    const int colB = nhalf * 128 + (lane & 15);
    const int rb = rowBase + hf * 8;
#pragma unroll
    for (int ntl = 0; ntl < 8; ++ntl) {
        const int col = colB + ntl * 16;
#pragma unroll
        for (int r = 0; r < 8; ++r) {
            const size_t idx = (size_t)(rb + r) * DIM + col;
            const float kv = fast_tanh(acc[ntl][r]);
            if (STAGE == 0) {
                kout[idx] = kv;
                accb[idx] = wgt * kv;
            } else if (STAGE == 1 || STAGE == 2) {
                kout[idx] = kv;
                accb[idx] += wgt * kv;
            } else {
                vout[idx] = v[idx] + accb[idx] + wgt * kv;
            }
        }
    }
}

extern "C" void kernel_launch(void* const* d_in, const int* in_sizes, int n_in,
                              void* d_out, int out_size, void* d_ws, size_t ws_size,
                              hipStream_t stream) {
    const float* z = (const float*)d_in[0];   // [64,1,256,256,2]
    const float* Win = (const float*)d_in[1]; // [256,256,2]
    const float* bin = (const float*)d_in[2]; // [256,256]
    const float* Wf = (const float*)d_in[3];  // [256,256]
    float* out = (float*)d_out;               // [64,1,256,256] == v[b*256+k][j]

    const size_t NELEM = (size_t)MROWS * DIM; // 4,194,304 floats (16 MB)
    float* v = (float*)d_ws;                  // state
    float* ka = v + NELEM;                    // k ping
    float* kb = ka + NELEM;                   // k pong
    unsigned* WfB = (unsigned*)(kb + NELEM);  // packed Wf, 128 KB
    float* accb = out;                        // RK4 accumulator lives in d_out

    init_v0_kernel<<<dim3(16, 16, 64), dim3(16, 16), 0, stream>>>(z, Win, bin, v);
    pack_wf_kernel<<<128, 256, 0, stream>>>(Wf, WfB);

    const float dt = 1.0f / (float)NSTEPS;
    const uint4* gB = (const uint4*)WfB;
    for (int s = 0; s < NSTEPS; ++s) {
        rk4_stage_kernel<0><<<256, 256, 0, stream>>>(v, nullptr, ka, accb, gB,
                                                     nullptr, 0.0f, dt / 6.0f);
        rk4_stage_kernel<1><<<256, 256, 0, stream>>>(v, ka, kb, accb, gB,
                                                     nullptr, 0.5f * dt, dt / 3.0f);
        rk4_stage_kernel<2><<<256, 256, 0, stream>>>(v, kb, ka, accb, gB,
                                                     nullptr, 0.5f * dt, dt / 3.0f);
        float* vo = (s == NSTEPS - 1) ? out : v;
        rk4_stage_kernel<3><<<256, 256, 0, stream>>>(v, ka, nullptr, accb, gB,
                                                     vo, dt, dt / 6.0f);
    }
}
